// WarpClassifier_87428354278163
// MI455X (gfx1250) — matrix operand order repaired and checked by reading, not compiled, not run on hardware
//
#include <hip/hip_runtime.h>

typedef __attribute__((ext_vector_type(16))) __bf16 v16bf;
typedef __attribute__((ext_vector_type(8)))  float  v8f;

#define TILE_M 128
#define TILE_N 128
#define TILE_K 32

#if defined(__HIP_DEVICE_COMPILE__) && __has_builtin(__builtin_amdgcn_global_load_async_to_lds_b128)
#define HAVE_ASYNC_LDS 1
#else
#define HAVE_ASYNC_LDS 0
#endif

union FragU  { v16bf v; uint4 q[2]; };
union PackU  { unsigned short u[16]; uint4 q[2]; };
union Pack8U { unsigned short u[8];  uint4 q; };

// Split fp32 into bf16 hi (truncate) + bf16 lo (residual): x ~= hi + lo, ~2^-16 rel err.
__device__ __forceinline__ void cvt_split(float x, unsigned short& hi, unsigned short& lo) {
  unsigned int xi = __float_as_uint(x);
  hi = (unsigned short)(xi >> 16);
  float hif = __uint_as_float(((unsigned int)hi) << 16);
  float res = x - hif;
  lo = (unsigned short)(__float_as_uint(res) >> 16);
}

// A-matrix 16-bit 16x32 fragment (ISA 7.12.2): lanes 0-15 hold K{0-7,16-23},
// lanes 16-31 hold K{8-15,24-31} of row (lane&15). Two 16B chunks from LDS.
__device__ __forceinline__ v16bf load_fragA(const unsigned short* base, int row, int lane) {
  const int k0 = (lane & 16) ? 8 : 0;
  const unsigned short* p = base + row * TILE_K + k0;
  FragU f;
  f.q[0] = *(const uint4*)(p);
  f.q[1] = *(const uint4*)(p + 16);
  return f.v;
}

// B-matrix 16-bit 32x16 fragment: lanes 0-15 hold K{0-7,16-23}, lanes 16-31 hold
// K{8-15,24-31} of column (lane&15). Two 16B chunks from LDS.
__device__ __forceinline__ v16bf load_fragB(const unsigned short* base, int row, int lane) {
  const int k0 = (lane & 16) ? 8 : 0;
  const unsigned short* p = base + row * TILE_K + k0;
  FragU f;
  f.q[0] = *(const uint4*)(p);
  f.q[1] = *(const uint4*)(p + 16);
  return f.v;
}

// 16B global->LDS copy: async (ASYNCcnt-tracked, no VGPR data) when available.
// Builtin prototype (from hipcc diagnostic): first param is
// `int __vector(4) __attribute__((address_space(1)))*` (global src), then LDS dst.
typedef int v4i_vs __attribute__((vector_size(16)));

__device__ __forceinline__ void cp16_to_lds(const unsigned short* g, unsigned short* l) {
#if HAVE_ASYNC_LDS
  using gv = __attribute__((address_space(1))) v4i_vs;
  using lv = __attribute__((address_space(3))) v4i_vs;
  __builtin_amdgcn_global_load_async_to_lds_b128((gv*)g, (lv*)l, 0, 0);
#else
  *(uint4*)l = *(const uint4*)g;
#endif
}

__device__ __forceinline__ void async_wait0() {
#if HAVE_ASYNC_LDS
#if __has_builtin(__builtin_amdgcn_s_wait_asynccnt)
  __builtin_amdgcn_s_wait_asynccnt(0);
#else
  asm volatile("s_wait_asynccnt 0x0" ::: "memory");
#endif
#endif
}

// ---------------------------------------------------------------------------
// Pre-split helper: f32 -> bf16 hi/lo planes. n8 = element_count / 8.
// ---------------------------------------------------------------------------
__global__ __launch_bounds__(256)
void split_f32(const float* __restrict__ src, unsigned short* __restrict__ hi,
               unsigned short* __restrict__ lo, int n8) {
  const int g = blockIdx.x * 256 + threadIdx.x;
  if (g >= n8) return;
  const float* s = src + (size_t)g * 8;
  const float4 a = *(const float4*)s;
  const float4 b = *(const float4*)(s + 4);
  const float v[8] = {a.x, a.y, a.z, a.w, b.x, b.y, b.z, b.w};
  Pack8U ph, pl;
#pragma unroll
  for (int i = 0; i < 8; ++i) cvt_split(v[i], ph.u[i], pl.u[i]);
  *(uint4*)(hi + (size_t)g * 8) = ph.q;
  *(uint4*)(lo + (size_t)g * 8) = pl.q;
}

// ---------------------------------------------------------------------------
// Fast GEMM: C[M,N] = (Ah+Al)[M,K] @ ((Bh+Bl)[N,K])^T + bias, operands pre-split
// bf16 hi/lo planes, K-contiguous. Async global->LDS tile copies, LDS double
// buffer, 24 v_wmma_f32_16x16x32_bf16 per wave per K-step (bf16x3).
// ---------------------------------------------------------------------------
__global__ __launch_bounds__(256)
void gemm_bf16x3_split(const unsigned short* __restrict__ Ah, const unsigned short* __restrict__ Al,
                       const unsigned short* __restrict__ Bh, const unsigned short* __restrict__ Bl,
                       const float* __restrict__ bias, float* __restrict__ C,
                       int M, int N, int K) {
  __shared__ __align__(16) unsigned short sAh[2][TILE_M * TILE_K];
  __shared__ __align__(16) unsigned short sAl[2][TILE_M * TILE_K];
  __shared__ __align__(16) unsigned short sBh[2][TILE_N * TILE_K];
  __shared__ __align__(16) unsigned short sBl[2][TILE_N * TILE_K];

  const int tid  = threadIdx.x;
  const int lane = tid & 31;
  const int wave = tid >> 5;      // 8 waves
  const int wm   = wave >> 2;     // 0..1  (64 rows each)
  const int wn   = wave & 3;      // 0..3  (32 cols each)
  const int bm   = blockIdx.x * TILE_M;
  const int bn   = blockIdx.y * TILE_N;

  v8f acc[4][2] = {};

  // Each of the 4 tile planes is 128 rows x 64B; 512 16B-chunks; 2 per thread.
  auto issue_tile = [&](int kt, int b) {
#pragma unroll
    for (int i = 0; i < 2; ++i) {
      const int chunk = tid * 2 + i;          // 0..511
      const int r  = chunk >> 2;              // tile row 0..127
      const int c8 = (chunk & 3) * 8;         // ushort col offset {0,8,16,24}
      const int lo = r * TILE_K + c8;         // LDS ushort offset
      cp16_to_lds(Ah + (size_t)(bm + r) * K + kt + c8, &sAh[b][lo]);
      cp16_to_lds(Al + (size_t)(bm + r) * K + kt + c8, &sAl[b][lo]);
      int nr = bn + r;
      if (nr > N - 1) nr = N - 1;             // clamp: junk cols n>=N never stored
      cp16_to_lds(Bh + (size_t)nr * K + kt + c8, &sBh[b][lo]);
      cp16_to_lds(Bl + (size_t)nr * K + kt + c8, &sBl[b][lo]);
    }
  };

  issue_tile(0, 0);
  int buf = 0;
  for (int kt = 0; kt < K; kt += TILE_K) {
    async_wait0();       // this wave's async copies into `buf` landed
    __syncthreads();     // all waves done writing `buf` and done reading `buf^1`
    if (kt + TILE_K < K) issue_tile(kt + TILE_K, buf ^ 1);  // overlaps WMMAs below

    v16bf ah[4], al[4], bh[2], bl[2];
#pragma unroll
    for (int mi = 0; mi < 4; ++mi) {
      const int r = wm * 64 + mi * 16 + (lane & 15);
      ah[mi] = load_fragA(&sAh[buf][0], r, lane);
      al[mi] = load_fragA(&sAl[buf][0], r, lane);
    }
#pragma unroll
    for (int ni = 0; ni < 2; ++ni) {
      const int r = wn * 32 + ni * 16 + (lane & 15);
      bh[ni] = load_fragB(&sBh[buf][0], r, lane);
      bl[ni] = load_fragB(&sBl[buf][0], r, lane);
    }
#pragma unroll
    for (int mi = 0; mi < 4; ++mi) {
#pragma unroll
      for (int ni = 0; ni < 2; ++ni) {
        acc[mi][ni] = __builtin_amdgcn_wmma_f32_16x16x32_bf16(
            false, ah[mi], false, bh[ni], (short)0, acc[mi][ni], false, false);
        acc[mi][ni] = __builtin_amdgcn_wmma_f32_16x16x32_bf16(
            false, al[mi], false, bh[ni], (short)0, acc[mi][ni], false, false);
        acc[mi][ni] = __builtin_amdgcn_wmma_f32_16x16x32_bf16(
            false, ah[mi], false, bl[ni], (short)0, acc[mi][ni], false, false);
      }
    }
    buf ^= 1;
  }

  // Epilogue: D layout -> VGPR r holds M=r (lanes 0-15) / M=r+8 (16-31), N=lane%16.
#pragma unroll
  for (int ni = 0; ni < 2; ++ni) {
    const int n = bn + wn * 32 + ni * 16 + (lane & 15);
    float bv = 0.0f;
    if (n < N) bv = bias[n];
#pragma unroll
    for (int mi = 0; mi < 4; ++mi) {
      const int mbase = bm + wm * 64 + mi * 16 + ((lane & 16) ? 8 : 0);
#pragma unroll
      for (int r = 0; r < 8; ++r) {
        if (n < N) C[(size_t)(mbase + r) * N + n] = acc[mi][ni][r] + bv;
      }
    }
  }
}

// ---------------------------------------------------------------------------
// Fallback GEMM: fp32 operands, in-loop bf16 hi/lo split into LDS.
// Used only when d_ws is too small for the pre-split planes.
// ---------------------------------------------------------------------------
__global__ __launch_bounds__(256)
void gemm_bf16x3_bias(const float* __restrict__ A, const float* __restrict__ B,
                      const float* __restrict__ bias, float* __restrict__ C,
                      int M, int N, int K) {
  __shared__ __align__(16) unsigned short As_hi[TILE_M * TILE_K];
  __shared__ __align__(16) unsigned short As_lo[TILE_M * TILE_K];
  __shared__ __align__(16) unsigned short Bs_hi[TILE_N * TILE_K];
  __shared__ __align__(16) unsigned short Bs_lo[TILE_N * TILE_K];

  const int tid  = threadIdx.x;
  const int lane = tid & 31;
  const int wave = tid >> 5;
  const int wm   = wave >> 2;
  const int wn   = wave & 3;
  const int bm   = blockIdx.x * TILE_M;
  const int bn   = blockIdx.y * TILE_N;

  v8f acc[4][2] = {};

  const int lrow = tid >> 1;
  const int lcol = (tid & 1) << 4;
  const float* Aptr = A + (size_t)(bm + lrow) * K + lcol;
  int nrow = bn + lrow;
  if (nrow > N - 1) nrow = N - 1;
  const float* Bptr = B + (size_t)nrow * K + lcol;

  float4 a_st[4], b_st[4];
#pragma unroll
  for (int i = 0; i < 4; ++i) {
    a_st[i] = *(const float4*)(Aptr + i * 4);
    b_st[i] = *(const float4*)(Bptr + i * 4);
  }

  for (int kt = 0; kt < K; kt += TILE_K) {
    {
      PackU ph, pl;
      const float* s = (const float*)a_st;
#pragma unroll
      for (int i = 0; i < 16; ++i) cvt_split(s[i], ph.u[i], pl.u[i]);
      *(uint4*)(&As_hi[lrow * TILE_K + lcol])     = ph.q[0];
      *(uint4*)(&As_hi[lrow * TILE_K + lcol + 8]) = ph.q[1];
      *(uint4*)(&As_lo[lrow * TILE_K + lcol])     = pl.q[0];
      *(uint4*)(&As_lo[lrow * TILE_K + lcol + 8]) = pl.q[1];
      const float* s2 = (const float*)b_st;
#pragma unroll
      for (int i = 0; i < 16; ++i) cvt_split(s2[i], ph.u[i], pl.u[i]);
      *(uint4*)(&Bs_hi[lrow * TILE_K + lcol])     = ph.q[0];
      *(uint4*)(&Bs_hi[lrow * TILE_K + lcol + 8]) = ph.q[1];
      *(uint4*)(&Bs_lo[lrow * TILE_K + lcol])     = pl.q[0];
      *(uint4*)(&Bs_lo[lrow * TILE_K + lcol + 8]) = pl.q[1];
    }
    __syncthreads();

    if (kt + TILE_K < K) {
#pragma unroll
      for (int i = 0; i < 4; ++i) {
        a_st[i] = *(const float4*)(Aptr + kt + TILE_K + i * 4);
        b_st[i] = *(const float4*)(Bptr + kt + TILE_K + i * 4);
      }
    }

    v16bf ah[4], al[4], bh[2], bl[2];
#pragma unroll
    for (int mi = 0; mi < 4; ++mi) {
      const int r = wm * 64 + mi * 16 + (lane & 15);
      ah[mi] = load_fragA(As_hi, r, lane);
      al[mi] = load_fragA(As_lo, r, lane);
    }
#pragma unroll
    for (int ni = 0; ni < 2; ++ni) {
      const int r = wn * 32 + ni * 16 + (lane & 15);
      bh[ni] = load_fragB(Bs_hi, r, lane);
      bl[ni] = load_fragB(Bs_lo, r, lane);
    }
#pragma unroll
    for (int mi = 0; mi < 4; ++mi) {
#pragma unroll
      for (int ni = 0; ni < 2; ++ni) {
        acc[mi][ni] = __builtin_amdgcn_wmma_f32_16x16x32_bf16(
            false, ah[mi], false, bh[ni], (short)0, acc[mi][ni], false, false);
        acc[mi][ni] = __builtin_amdgcn_wmma_f32_16x16x32_bf16(
            false, al[mi], false, bh[ni], (short)0, acc[mi][ni], false, false);
        acc[mi][ni] = __builtin_amdgcn_wmma_f32_16x16x32_bf16(
            false, ah[mi], false, bl[ni], (short)0, acc[mi][ni], false, false);
      }
    }
    __syncthreads();
  }

#pragma unroll
  for (int ni = 0; ni < 2; ++ni) {
    const int n = bn + wn * 32 + ni * 16 + (lane & 15);
    float bv = 0.0f;
    if (n < N) bv = bias[n];
#pragma unroll
    for (int mi = 0; mi < 4; ++mi) {
      const int mbase = bm + wm * 64 + mi * 16 + ((lane & 16) ? 8 : 0);
#pragma unroll
      for (int r = 0; r < 8; ++r) {
        if (n < N) C[(size_t)(mbase + r) * N + n] = acc[mi][ni][r] + bv;
      }
    }
  }
}

// ---------------------------------------------------------------------------
// One block per row: softmax -> cumsum -> uniform-grid linear interp.
// split_out=1: emit warped as bf16 hi/lo planes (fast path).
// split_out=0: overwrite logits in-place with warped f32 (fallback path).
// ---------------------------------------------------------------------------
__global__ __launch_bounds__(256)
void softmax_cumsum_interp(const float* __restrict__ X, float* __restrict__ L,
                           unsigned short* __restrict__ Wh, unsigned short* __restrict__ Wl,
                           int split_out) {
  const int NI = 2048;
  __shared__ float x_s[2048];
  __shared__ float red[256];
  const int row = blockIdx.x;
  const int t = threadIdx.x;
  const float* Lr = L + (size_t)row * NI;
  const float* Xr = X + (size_t)row * NI;

  float v[8];
  {
    float4 l0 = *(const float4*)(Lr + t * 8);
    float4 l1 = *(const float4*)(Lr + t * 8 + 4);
    v[0] = l0.x; v[1] = l0.y; v[2] = l0.z; v[3] = l0.w;
    v[4] = l1.x; v[5] = l1.y; v[6] = l1.z; v[7] = l1.w;
  }
#pragma unroll
  for (int k = 0; k < 8; ++k) x_s[t + k * 256] = Xr[t + k * 256];

  float m = v[0];
#pragma unroll
  for (int i = 1; i < 8; ++i) m = fmaxf(m, v[i]);
  red[t] = m;
  __syncthreads();
  for (int off = 128; off > 0; off >>= 1) {
    if (t < off) red[t] = fmaxf(red[t], red[t + off]);
    __syncthreads();
  }
  m = red[0];
  __syncthreads();

  float pref[8];
  float run = 0.0f;
#pragma unroll
  for (int i = 0; i < 8; ++i) { run += __expf(v[i] - m); pref[i] = run; }

  red[t] = run;
  __syncthreads();
  for (int off = 1; off < 256; off <<= 1) {
    float add = 0.0f;
    if (t >= off) add = red[t - off];
    __syncthreads();
    red[t] += add;
    __syncthreads();
  }
  const float excl = red[t] - run;
  const float invS = 1.0f / red[255];

  float outv[8];
#pragma unroll
  for (int i = 0; i < 8; ++i) {
    float g = (excl + pref[i]) * invS;            // gamma in (0, 1]
    float tc = g * 2047.0f;
    tc = fminf(fmaxf(tc, 0.0f), 2047.0f);         // jnp.interp end clamping
    int i0 = (int)tc;
    if (i0 > 2046) i0 = 2046;
    const float fr = tc - (float)i0;
    const float y0 = x_s[i0], y1 = x_s[i0 + 1];
    outv[i] = y0 + fr * (y1 - y0);
  }

  if (split_out) {
    Pack8U ph, pl;
#pragma unroll
    for (int i = 0; i < 8; ++i) cvt_split(outv[i], ph.u[i], pl.u[i]);
    *(uint4*)(Wh + (size_t)row * NI + t * 8) = ph.q;
    *(uint4*)(Wl + (size_t)row * NI + t * 8) = pl.q;
  } else {
    float* Lw = L + (size_t)row * NI;
    *(float4*)(Lw + t * 8)     = make_float4(outv[0], outv[1], outv[2], outv[3]);
    *(float4*)(Lw + t * 8 + 4) = make_float4(outv[4], outv[5], outv[6], outv[7]);
  }
}

extern "C" void kernel_launch(void* const* d_in, const int* in_sizes, int n_in,
                              void* d_out, int out_size, void* d_ws, size_t ws_size,
                              hipStream_t stream) {
  (void)in_sizes; (void)n_in; (void)out_size;
  const float* ts = (const float*)d_in[0];   // [16384, 2048]
  const float* ww = (const float*)d_in[1];   // [2048, 2048]
  const float* wb = (const float*)d_in[2];   // [2048]
  const float* fw = (const float*)d_in[3];   // [100, 2048]
  const float* fb = (const float*)d_in[4];   // [100]
  float* out = (float*)d_out;                // [16384, 100]

  const int Bsz = 16384, NI = 2048, NC = 100;
  dim3 blk(256);
  dim3 g1(Bsz / TILE_M, NI / TILE_N);                  // 128 x 16
  dim3 g3(Bsz / TILE_M, (NC + TILE_N - 1) / TILE_N);   // 128 x 1

  // Workspace layout (fast path):
  //   logits f32            : 16384*2048*4 = 128 MB
  //   x/warped hi, lo planes: 2 * 64 MB   (X split for GEMM1, then warped split)
  //   W hi, lo planes       : 2 * 8 MB
  //   fc hi, lo planes      : 2 * 0.4 MB
  char* ws = (char*)d_ws;
  size_t off = 0;
  float* logits = (float*)(ws + off);          off += (size_t)Bsz * NI * 4;
  unsigned short* xh = (unsigned short*)(ws + off); off += (size_t)Bsz * NI * 2;
  unsigned short* xl = (unsigned short*)(ws + off); off += (size_t)Bsz * NI * 2;
  unsigned short* wh = (unsigned short*)(ws + off); off += (size_t)NI * NI * 2;
  unsigned short* wl = (unsigned short*)(ws + off); off += (size_t)NI * NI * 2;
  unsigned short* fh = (unsigned short*)(ws + off); off += (size_t)NC * NI * 2;
  unsigned short* fl = (unsigned short*)(ws + off); off += (size_t)NC * NI * 2;
  const size_t required = off;

  if (ws_size >= required) {
    // Pre-split all GEMM operands once (bandwidth-trivial).
    const int n8x = Bsz * NI / 8, n8w = NI * NI / 8, n8f = NC * NI / 8;
    split_f32<<<dim3((n8x + 255) / 256), blk, 0, stream>>>(ts, xh, xl, n8x);
    split_f32<<<dim3((n8w + 255) / 256), blk, 0, stream>>>(ww, wh, wl, n8w);
    split_f32<<<dim3((n8f + 255) / 256), blk, 0, stream>>>(fw, fh, fl, n8f);

    gemm_bf16x3_split<<<g1, blk, 0, stream>>>(xh, xl, wh, wl, wb, logits, Bsz, NI, NI);
    // Warped written as hi/lo planes over the (now consumed) X planes.
    softmax_cumsum_interp<<<Bsz, blk, 0, stream>>>(ts, logits, xh, xl, 1);
    gemm_bf16x3_split<<<g3, blk, 0, stream>>>(xh, xl, fh, fl, fb, out, Bsz, NC, NI);
  } else {
    // Small-workspace fallback: in-kernel split, warped overwrites logits.
    gemm_bf16x3_bias<<<g1, blk, 0, stream>>>(ts, ww, wb, logits, Bsz, NI, NI);
    softmax_cumsum_interp<<<Bsz, blk, 0, stream>>>(ts, logits, nullptr, nullptr, 0);
    gemm_bf16x3_bias<<<g3, blk, 0, stream>>>(logits, fw, fb, out, Bsz, NC, NI);
  }
}
